// FullAttention_82059645157871
// MI455X (gfx1250) — matrix operand order repaired and checked by reading, not compiled, not run on hardware
//
#include <hip/hip_runtime.h>

typedef __attribute__((ext_vector_type(16))) _Float16 v16h;
typedef __attribute__((ext_vector_type(8)))  _Float16 v8h;
typedef __attribute__((ext_vector_type(2)))  _Float16 v2h;
typedef __attribute__((ext_vector_type(8)))  float    v8f;

#define T_SEQ    2048
#define DHEAD    64
#define BLOCK_M  128
#define BLOCK_N  64
#define NWAVES   8
#define KVSTRIDE 72   // halves per row (64 + 8 pad, 144B = 16B aligned) for sK / sVT
#define SSTRIDE  68   // floats per row (64 + 4 pad, 272B = 16B aligned) for sS

__device__ inline v8h cvt8(float4 a, float4 b, float s) {
    v8h r;
    r[0] = (_Float16)(a.x * s); r[1] = (_Float16)(a.y * s);
    r[2] = (_Float16)(a.z * s); r[3] = (_Float16)(a.w * s);
    r[4] = (_Float16)(b.x * s); r[5] = (_Float16)(b.y * s);
    r[6] = (_Float16)(b.z * s); r[7] = (_Float16)(b.w * s);
    return r;
}

// load one 32x16 f16 B-operand (two 8-half runs at p and p+16 per lane) from LDS
__device__ inline v16h load_bop(const _Float16* p) {
    v16h b;
    v8h* bp = (v8h*)&b;
    bp[0] = *(const v8h*)p;
    bp[1] = *(const v8h*)(p + 16);
    return b;
}

__global__ __launch_bounds__(256)
void fa_causal_fwd(const float* __restrict__ Q, const float* __restrict__ K,
                   const float* __restrict__ V, float* __restrict__ O)
{
    __shared__ _Float16 sK [2][64 * KVSTRIDE];          // key-major, f16
    __shared__ _Float16 sVT[2][64 * KVSTRIDE];          // dim-major (transposed V), f16
    __shared__ float    sS [NWAVES][16 * SSTRIDE];      // per-wave raw scores, f32

    const int mb   = blockIdx.x;           // 0..15 : 128-row Q block
    const int bh   = blockIdx.y;           // 0..31 : (b,h)
    const int tid  = threadIdx.x;
    const int wv   = tid >> 5;
    const int lane = tid & 31;
    const int h    = lane >> 4;
    const int ln   = lane & 15;

    const int    base_m = mb * BLOCK_M + wv * 16;
    const size_t bhT    = (size_t)bh * T_SEQ;
    // 1/sqrt(64) * log2(e): S is produced directly in the exp2 domain
    const float  scale  = 0.125f * 1.4426950408889634f;

    // K staging coords: 256 threads cover 64x64, 16 contiguous dims each
    const int srow = tid >> 2;             // key row 0..63
    const int sc0  = (tid & 3) * 16;       // dim 0,16,32,48
    // V staging coords: each thread owns 2 adjacent keys x 8 dims (packed stores)
    const int vk2  = (tid & 31) * 2;       // key pair 0,2,..,62
    const int vd0  = (tid >> 5) * 8;       // dim group 0,8,..,56

    // ---- Q tile (16x64) -> two 16x32 f16 A-operands (ISA A-layout) ----
    v16h qa0, qa1;
    {
        const float* qrow = Q + (bhT + base_m + ln) * DHEAD;
        const float* p1 = qrow + h * 8;
        const float* p2 = qrow + 16 + h * 8;
        const float* p3 = qrow + 32 + h * 8;
        const float* p4 = qrow + 48 + h * 8;
        v8h* q0 = (v8h*)&qa0;
        q0[0] = cvt8(*(const float4*)p1, *(const float4*)(p1 + 4), scale);
        q0[1] = cvt8(*(const float4*)p2, *(const float4*)(p2 + 4), scale);
        v8h* q1 = (v8h*)&qa1;
        q1[0] = cvt8(*(const float4*)p3, *(const float4*)(p3 + 4), scale);
        q1[1] = cvt8(*(const float4*)p4, *(const float4*)(p4 + 4), scale);
    }

    v8f acc[4];
    #pragma unroll
    for (int t = 0; t < 4; ++t)
        #pragma unroll
        for (int r = 0; r < 8; ++r) acc[t][r] = 0.0f;
    float m_l = -3.0e38f;                  // per-lane row running max (log2 domain)
    float l_l = 0.0f;                      // per-lane row running sum

    const int kb_end = 2 * mb + 2;         // 64-key blocks covering [0, mb*128+128)

    // ---- prologue: stage key block 0 into buffer 0 ----
    {
        const float* kp = K + (bhT + srow) * DHEAD + sc0;
        float4 k0 = ((const float4*)kp)[0], k1 = ((const float4*)kp)[1];
        float4 k2 = ((const float4*)kp)[2], k3 = ((const float4*)kp)[3];
        *(v8h*)&sK[0][srow * KVSTRIDE + sc0]     = cvt8(k0, k1, 1.0f);
        *(v8h*)&sK[0][srow * KVSTRIDE + sc0 + 8] = cvt8(k2, k3, 1.0f);

        const float* vp0 = V + (bhT + vk2) * DHEAD + vd0;
        const float* vp1 = vp0 + DHEAD;
        float4 a0 = ((const float4*)vp0)[0], a1 = ((const float4*)vp0)[1];
        float4 b0 = ((const float4*)vp1)[0], b1 = ((const float4*)vp1)[1];
        float va[8] = {a0.x, a0.y, a0.z, a0.w, a1.x, a1.y, a1.z, a1.w};
        float vb[8] = {b0.x, b0.y, b0.z, b0.w, b1.x, b1.y, b1.z, b1.w};
        #pragma unroll
        for (int i = 0; i < 8; ++i) {
            v2h t; t[0] = (_Float16)va[i]; t[1] = (_Float16)vb[i];
            *(v2h*)&sVT[0][(vd0 + i) * KVSTRIDE + vk2] = t;
        }
    }
    __syncthreads();

    for (int kb = 0; kb < kb_end; ++kb) {
        const int  cur = kb & 1, nxt = cur ^ 1;
        const bool have_next = (kb + 1 < kb_end);

        // ---- issue next block's global loads early (hidden behind compute) ----
        float4 nk0, nk1, nk2, nk3, nva0, nva1, nvb0, nvb1;
        if (have_next) {
            const float* kp = K + (bhT + (size_t)(kb + 1) * BLOCK_N + srow) * DHEAD + sc0;
            nk0 = ((const float4*)kp)[0]; nk1 = ((const float4*)kp)[1];
            nk2 = ((const float4*)kp)[2]; nk3 = ((const float4*)kp)[3];
            const float* vp0 = V + (bhT + (size_t)(kb + 1) * BLOCK_N + vk2) * DHEAD + vd0;
            const float* vp1 = vp0 + DHEAD;
            nva0 = ((const float4*)vp0)[0]; nva1 = ((const float4*)vp0)[1];
            nvb0 = ((const float4*)vp1)[0]; nvb1 = ((const float4*)vp1)[1];
            if (kb + 2 < kb_end) {
                __builtin_prefetch(kp + BLOCK_N * DHEAD, 0, 1);
                __builtin_prefetch(vp0 + BLOCK_N * DHEAD, 0, 1);
            }
        }

        // ---- compute (wave-uniform guard keeps EXEC all-ones for WMMA) ----
        if (kb * BLOCK_N <= base_m + 15) {
            const _Float16* sKb = sK[cur];
            const _Float16* sVb = sVT[cur];
            float*          sSw = sS[wv];
            const _Float16* krow = &sKb[ln * KVSTRIDE];
            const _Float16* vrow = &sVb[ln * KVSTRIDE];

            // S = Q @ K^T : batch 4 B-operand loads, then 4 WMMAs (per K half)
            v8f s[4];
            {
                v16h kb0[4];
                #pragma unroll
                for (int nt = 0; nt < 4; ++nt)
                    kb0[nt] = load_bop(krow + nt * 16 * KVSTRIDE + h * 8);
                v8f z;
                #pragma unroll
                for (int r = 0; r < 8; ++r) z[r] = 0.0f;   // folds to inline C=0
                #pragma unroll
                for (int nt = 0; nt < 4; ++nt)
                    s[nt] = __builtin_amdgcn_wmma_f32_16x16x32_f16(
                                false, qa0, false, kb0[nt], (short)0, z,
                                false, false);
            }
            {
                v16h kb1[4];
                #pragma unroll
                for (int nt = 0; nt < 4; ++nt)
                    kb1[nt] = load_bop(krow + nt * 16 * KVSTRIDE + 32 + h * 8);
                #pragma unroll
                for (int nt = 0; nt < 4; ++nt)
                    s[nt] = __builtin_amdgcn_wmma_f32_16x16x32_f16(
                                false, qa1, false, kb1[nt], (short)0, s[nt],
                                false, false);
            }
            // spill raw C tiles to per-wave LDS strip
            #pragma unroll
            for (int nt = 0; nt < 4; ++nt)
                #pragma unroll
                for (int r = 0; r < 8; ++r)
                    sSw[(r + 8 * h) * SSTRIDE + nt * 16 + ln] = s[nt][r];

            // ---- read back row-per-lane (exactly the PV A-operand positions) ----
            float xv[32];
            const float* srd = &sSw[ln * SSTRIDE + h * 8];
            #pragma unroll
            for (int q = 0; q < 4; ++q) {
                float4 a = *(const float4*)(srd + q * 16);
                float4 b = *(const float4*)(srd + q * 16 + 4);
                xv[q * 8 + 0] = a.x; xv[q * 8 + 1] = a.y;
                xv[q * 8 + 2] = a.z; xv[q * 8 + 3] = a.w;
                xv[q * 8 + 4] = b.x; xv[q * 8 + 5] = b.y;
                xv[q * 8 + 6] = b.z; xv[q * 8 + 7] = b.w;
            }

            // PV B-operands (first key half): in flight during the softmax VALU
            v16h vb0[4];
            #pragma unroll
            for (int nb = 0; nb < 4; ++nb)
                vb0[nb] = load_bop(vrow + nb * 16 * KVSTRIDE + h * 8);

            // causal mask (row = base_m + ln, key = kb*64 + q*16 + h*8 + i)
            if (kb * BLOCK_N + BLOCK_N - 1 > base_m) {
                const int thr2 = base_m + ln - kb * BLOCK_N - h * 8;
                #pragma unroll
                for (int e = 0; e < 32; ++e) {
                    const int jc = (e >> 3) * 16 + (e & 7);
                    if (jc > thr2) xv[e] = -3.0e38f;
                }
            }

            // row max: register tree + one cross-half exchange
            float mx = xv[0];
            #pragma unroll
            for (int e = 1; e < 32; ++e) mx = fmaxf(mx, xv[e]);
            mx = fmaxf(mx, __shfl_xor(mx, 16, 32));
            const float mnew = fmaxf(m_l, mx);
            const float al   = __builtin_amdgcn_exp2f(m_l - mnew);
            m_l = mnew;

            // exp2 -> directly the two PV A-operands; accumulate row sum
            v16h pa0, pa1;
            float rs = 0.0f;
            #pragma unroll
            for (int e = 0; e < 32; ++e) {
                const float p = __builtin_amdgcn_exp2f(xv[e] - mnew);
                rs += p;
                if (e < 16) pa0[e] = (_Float16)p;
                else        pa1[e - 16] = (_Float16)p;
            }
            rs += __shfl_xor(rs, 16, 32);
            l_l = l_l * al + rs;

            // rescale accumulators (alpha broadcast: lane layout -> C layout)
            #pragma unroll
            for (int r = 0; r < 8; ++r) {
                const float ac = __shfl(al, r + 8 * h, 32);
                acc[0][r] *= ac; acc[1][r] *= ac;
                acc[2][r] *= ac; acc[3][r] *= ac;
            }

            // O += P @ V : first key half, then second (loads behind WMMAs)
            #pragma unroll
            for (int nb = 0; nb < 4; ++nb)
                acc[nb] = __builtin_amdgcn_wmma_f32_16x16x32_f16(
                              false, pa0, false, vb0[nb], (short)0, acc[nb],
                              false, false);
            v16h vb1[4];
            #pragma unroll
            for (int nb = 0; nb < 4; ++nb)
                vb1[nb] = load_bop(vrow + nb * 16 * KVSTRIDE + 32 + h * 8);
            #pragma unroll
            for (int nb = 0; nb < 4; ++nb)
                acc[nb] = __builtin_amdgcn_wmma_f32_16x16x32_f16(
                              false, pa1, false, vb1[nb], (short)0, acc[nb],
                              false, false);
        }

        // ---- write next block into the other buffer, then one barrier ----
        if (have_next) {
            *(v8h*)&sK[nxt][srow * KVSTRIDE + sc0]     = cvt8(nk0, nk1, 1.0f);
            *(v8h*)&sK[nxt][srow * KVSTRIDE + sc0 + 8] = cvt8(nk2, nk3, 1.0f);
            float va[8] = {nva0.x, nva0.y, nva0.z, nva0.w,
                           nva1.x, nva1.y, nva1.z, nva1.w};
            float vb[8] = {nvb0.x, nvb0.y, nvb0.z, nvb0.w,
                           nvb1.x, nvb1.y, nvb1.z, nvb1.w};
            #pragma unroll
            for (int i = 0; i < 8; ++i) {
                v2h t; t[0] = (_Float16)va[i]; t[1] = (_Float16)vb[i];
                *(v2h*)&sVT[nxt][(vd0 + i) * KVSTRIDE + vk2] = t;
            }
        }
        __syncthreads();
    }

    // ---- normalize (l broadcast lane->C layout) and store ----
    float* orow = O + (bhT + base_m) * DHEAD;
    #pragma unroll
    for (int r = 0; r < 8; ++r) {
        const float lr  = __shfl(l_l, r + 8 * h, 32);
        const float inv = 1.0f / lr;
        #pragma unroll
        for (int t = 0; t < 4; ++t)
            orow[(size_t)(r + 8 * h) * DHEAD + t * 16 + ln] = acc[t][r] * inv;
    }
}

extern "C" void kernel_launch(void* const* d_in, const int* in_sizes, int n_in,
                              void* d_out, int out_size, void* d_ws, size_t ws_size,
                              hipStream_t stream) {
    const float* q = (const float*)d_in[0];
    const float* k = (const float*)d_in[1];
    const float* v = (const float*)d_in[2];
    float* o = (float*)d_out;
    dim3 grid(T_SEQ / BLOCK_M, 32, 1);   // 16 Q-blocks x (B*H = 32)
    dim3 block(256, 1, 1);               // 8 waves of 32
    fa_causal_fwd<<<grid, block, 0, stream>>>(q, k, v, o);
}
